// MultiHeadAttention_52982716563963
// MI455X (gfx1250) — matrix operand order repaired and checked by reading, not compiled, not run on hardware
//
#include <hip/hip_runtime.h>
#include <hip/hip_bf16.h>

// ---------------------------------------------------------------------------
// MultiHeadAttention (RoPE + causal sliding-window) for gfx1250 / MI455X.
// Matmuls via v_wmma_f32_16x16x32_bf16 (f32 accumulate); GEMM A-tiles staged
// global->LDS with global_load_async_to_lds_b128 (ASYNCcnt, double-buffered).
// ---------------------------------------------------------------------------

typedef __attribute__((ext_vector_type(16))) __bf16 v16bf;
typedef __attribute__((ext_vector_type(8)))  __bf16 bf16x8;
typedef __attribute__((ext_vector_type(8)))  float  v8f;

union FragAB { v16bf v; bf16x8 h[2]; };

constexpr int D_MODEL   = 2048;
constexpr int NUM_HEADS = 16;
constexpr int HEAD_DIM  = 128;
constexpr int WINDOW    = 1024;
constexpr int BATCH     = 2;
constexpr int SEQL      = 2048;
constexpr int MROWS     = BATCH * SEQL;          // 4096
constexpr float QSCALE  = 0.08838834764831845f;  // 1/sqrt(HEAD_DIM)

// ------------------------------- f32 -> bf16 -------------------------------
__global__ void cvt_f32_bf16(const float* __restrict__ s, __bf16* __restrict__ d, int n) {
    int i = blockIdx.x * blockDim.x + threadIdx.x;
    if (i < n) d[i] = (__bf16)s[i];
}

// Low 32 bits of a generic pointer into LDS == wave-relative LDS byte offset
// (see ISA 10.2).
__device__ __forceinline__ unsigned lds_off(const void* p) {
    return (unsigned)(unsigned long long)p;
}

// ------------------------- WMMA bf16 GEMM (C = A @ B) ----------------------
// A: [M,K] bf16 row-major.  B: [K,N] bf16 row-major.
// Block tile 128x128, BK=32, 256 threads = 8 waves (4x2), wave tile 32x64.
// Double-buffered LDS; A tile moved with global_load_async_to_lds_b128.
template<bool WRITE_F32>
__global__ __launch_bounds__(256) void gemm_bf16(
    const __bf16* __restrict__ A, const __bf16* __restrict__ Bw,
    __bf16* __restrict__ Cb, float* __restrict__ Cf,
    int M, int N, int K)
{
    constexpr int BM = 128, BN = 128, BK = 32;
    __shared__ __bf16 As[2][BM][BK];   // row-major A tiles (async-filled)
    __shared__ __bf16 Bs[2][BN][BK];   // B tiles stored transposed: Bs[..][n][k]

    const int tid  = threadIdx.x;
    const int lane = tid & 31;
    const int wave = tid >> 5;
    const int half = lane >> 4;
    const int l15  = lane & 15;

    const int bm = blockIdx.y * BM;
    const int bn = blockIdx.x * BN;
    const int wm = (wave & 3) * 32;   // wave row offset in tile
    const int wn = (wave >> 2) * 64;  // wave col offset in tile

    // global-load assignments
    const int ar = tid >> 1, ac = (tid & 1) * 16;  // A: 2 thr/row, 2x 8-elem vectors
    const int br = tid >> 3, bc = (tid & 7) * 16;  // B: 8 thr/row of 128

    const v8f vzero = {0.f,0.f,0.f,0.f,0.f,0.f,0.f,0.f};
    v8f acc[2][4];
#pragma unroll
    for (int i = 0; i < 2; ++i)
#pragma unroll
        for (int j = 0; j < 4; ++j) acc[i][j] = vzero;

    // ---- stage one 128x32 A slab + 32x128 B slab into buffer `buf` ----
    auto stage = [&](int k0, int buf) {
        // A: global -> LDS directly, async (2x b128 per thread)
        const __bf16* ag   = A + (size_t)(bm + ar) * K + (k0 + ac);
        const unsigned la0 = lds_off(&As[buf][ar][ac]);
        const unsigned la1 = la0 + 16u;
        const __bf16* ag1  = ag + 8;
        asm volatile("global_load_async_to_lds_b128 %0, %1, off"
                     :: "v"(la0), "v"(ag) : "memory");
        asm volatile("global_load_async_to_lds_b128 %0, %1, off"
                     :: "v"(la1), "v"(ag1) : "memory");
        // B: VGPR staging + transposed store (fragment reads become contiguous)
        const __bf16* bg = Bw + (size_t)(k0 + br) * N + (bn + bc);
        bf16x8 b0 = *(const bf16x8*)(bg);
        bf16x8 b1 = *(const bf16x8*)(bg + 8);
#pragma unroll
        for (int j = 0; j < 8; ++j) {
            Bs[buf][bc + j][br]     = b0[j];
            Bs[buf][bc + 8 + j][br] = b1[j];
        }
        __builtin_prefetch(bg + (size_t)BK * N, 0, 1);   // global_prefetch_b8
    };

    const int nk = K / BK;
    stage(0, 0);
    asm volatile("s_wait_asynccnt 0" ::: "memory");
    __syncthreads();

    for (int t = 0; t < nk; ++t) {
        const int cur = t & 1;
        if (t + 1 < nk) stage((t + 1) * BK, (t + 1) & 1);

        // ---- fragments from buffer `cur` ----
        FragAB af[2], bfr[4];
#pragma unroll
        for (int i = 0; i < 2; ++i) {
            // A 16x32 bf16 layout: lanes 0-15 rows, K 0..15; lanes16-31 K 16..31
            const __bf16* p = &As[cur][wm + i * 16 + l15][half * 16];
            af[i].h[0] = *(const bf16x8*)(p);
            af[i].h[1] = *(const bf16x8*)(p + 8);
        }
#pragma unroll
        for (int j = 0; j < 4; ++j) {
            // B 32x16 bf16 layout: lane n holds column n, K = half*16 .. +15
            const __bf16* p = &Bs[cur][wn + j * 16 + l15][half * 16];
            bfr[j].h[0] = *(const bf16x8*)(p);
            bfr[j].h[1] = *(const bf16x8*)(p + 8);
        }
#pragma unroll
        for (int i = 0; i < 2; ++i)
#pragma unroll
            for (int j = 0; j < 4; ++j)
                acc[i][j] = __builtin_amdgcn_wmma_f32_16x16x32_bf16(
                    false, af[i].v, false, bfr[j].v, (short)0, acc[i][j], false, false);

        // async fill of buffer (t+1)&1 must complete before next iteration reads it
        asm volatile("s_wait_asynccnt 0" ::: "memory");
        __syncthreads();
    }

    // ---- store C: VGPR r holds rows r (lanes 0-15) and r+8 (lanes 16-31), col = l15 ----
#pragma unroll
    for (int i = 0; i < 2; ++i)
#pragma unroll
        for (int j = 0; j < 4; ++j) {
            const int gn = bn + wn + j * 16 + l15;
#pragma unroll
            for (int r = 0; r < 8; ++r) {
                const int gm = bm + wm + i * 16 + r + 8 * half;
                const float v = acc[i][j][r];
                if (WRITE_F32) Cf[(size_t)gm * N + gn] = v;
                else           Cb[(size_t)gm * N + gn] = (__bf16)v;
            }
        }
}

// ----------------- RoPE (+score scale on Q) and attention layouts ----------
// qlin/klin/vlin: [B*S, D] bf16.  Outputs:
//   qh, kh: [B, H, S, hd] bf16 (Q pre-scaled by 1/sqrt(hd))
//   vt:     [B, H, hd, S] bf16 (transposed for PV WMMA B-operand)
__global__ __launch_bounds__(256) void rope_relayout(
    const __bf16* __restrict__ qlin, const __bf16* __restrict__ klin,
    const __bf16* __restrict__ vlin,
    const float* __restrict__ fcos, const float* __restrict__ fsin,
    __bf16* __restrict__ qh, __bf16* __restrict__ kh, __bf16* __restrict__ vt)
{
    const int tok = blockIdx.x;          // b*SEQ + s
    const int b = tok / SEQL, s = tok % SEQL;
    for (int p = threadIdx.x; p < D_MODEL / 2; p += 256) {
        const int hIdx = p >> 6;         // head
        const int i    = p & 63;         // rotary pair
        const size_t src = (size_t)tok * D_MODEL + hIdx * HEAD_DIM + 2 * i;
        const float c  = fcos[(size_t)s * 64 + i];
        const float sn = fsin[(size_t)s * 64 + i];

        const float qa = (float)qlin[src], qb = (float)qlin[src + 1];
        const float ka = (float)klin[src], kb = (float)klin[src + 1];

        const size_t dst = (((size_t)(b * NUM_HEADS + hIdx)) * SEQL + s) * HEAD_DIM + 2 * i;
        qh[dst]     = (__bf16)((qa * c - qb * sn) * QSCALE);
        qh[dst + 1] = (__bf16)((qa * sn + qb * c) * QSCALE);
        kh[dst]     = (__bf16)(ka * c - kb * sn);
        kh[dst + 1] = (__bf16)(ka * sn + kb * c);

        const size_t vbase = ((size_t)(b * NUM_HEADS + hIdx)) * HEAD_DIM * SEQL;
        vt[vbase + (size_t)(2 * i)     * SEQL + s] = vlin[src];
        vt[vbase + (size_t)(2 * i + 1) * SEQL + s] = vlin[src + 1];
    }
}

// ----------------------- flash-style windowed attention --------------------
// grid.x = B*H*(S/128); 256 threads = 8 waves; each wave owns 16 query rows.
__global__ __launch_bounds__(256) void attention_kernel(
    const __bf16* __restrict__ qh, const __bf16* __restrict__ kh,
    const __bf16* __restrict__ vt, __bf16* __restrict__ out)
{
    __shared__ __bf16 Ps[8][16][32];     // per-wave P scratch (C-layout -> A-layout)

    const int blk  = blockIdx.x;
    const int qblk = blk & 15;           // S/128 = 16
    const int bh   = blk >> 4;
    const int b = bh >> 4, h = bh & 15;

    const __bf16* Q  = qh + (size_t)bh * SEQL * HEAD_DIM;
    const __bf16* Kp = kh + (size_t)bh * SEQL * HEAD_DIM;
    const __bf16* Vt = vt + (size_t)bh * HEAD_DIM * SEQL;   // [hd][S]

    const int lane = threadIdx.x & 31;
    const int wave = threadIdx.x >> 5;
    const int half = lane >> 4;
    const int l15  = lane & 15;
    const int q0   = qblk * 128 + wave * 16;

    // persistent Q fragments: A 16x128 as 4x (16x32)
    FragAB qf[4];
#pragma unroll
    for (int c = 0; c < 4; ++c) {
        const __bf16* qp = Q + (size_t)(q0 + l15) * HEAD_DIM + c * 32 + half * 16;
        qf[c].h[0] = *(const bf16x8*)(qp);
        qf[c].h[1] = *(const bf16x8*)(qp + 8);
    }

    const v8f vzero = {0.f,0.f,0.f,0.f,0.f,0.f,0.f,0.f};
    v8f o[8];
#pragma unroll
    for (int j = 0; j < 8; ++j) o[j] = vzero;
    float mrun[8], lrun[8];
#pragma unroll
    for (int r = 0; r < 8; ++r) { mrun[r] = -1.0e30f; lrun[r] = 0.0f; }

    int kstart = qblk * 128 - WINDOW + 1;
    if (kstart < 0) kstart = 0;
    kstart &= ~31;                        // align to 32-key chunk
    const int kend = qblk * 128 + 128;    // causal upper bound for this block

    for (int kk = kstart; kk < kend; kk += 32) {
        // ---- scores: two 16x16 tiles over 32 keys, accumulate over hd=128 ----
        v8f sc0 = vzero, sc1 = vzero;
#pragma unroll
        for (int c = 0; c < 4; ++c) {
            FragAB kf0, kf1;   // B operand: lane = key column, rows = hd-chunk (contiguous in kh)
            const __bf16* kp0 = Kp + (size_t)(kk + l15)      * HEAD_DIM + c * 32 + half * 16;
            const __bf16* kp1 = Kp + (size_t)(kk + 16 + l15) * HEAD_DIM + c * 32 + half * 16;
            kf0.h[0] = *(const bf16x8*)(kp0);
            kf0.h[1] = *(const bf16x8*)(kp0 + 8);
            kf1.h[0] = *(const bf16x8*)(kp1);
            kf1.h[1] = *(const bf16x8*)(kp1 + 8);
            sc0 = __builtin_amdgcn_wmma_f32_16x16x32_bf16(false, qf[c].v, false, kf0.v, (short)0, sc0, false, false);
            sc1 = __builtin_amdgcn_wmma_f32_16x16x32_bf16(false, qf[c].v, false, kf1.v, (short)0, sc1, false, false);
        }

        // ---- mask + online softmax (row r lives in VGPR r, half picks +8) ----
#pragma unroll
        for (int r = 0; r < 8; ++r) {
            const int qrow = q0 + r + 8 * half;
            const int k0i = kk + l15, k1i = kk + 16 + l15;
            float s0 = sc0[r], s1 = sc1[r];
            const bool v0 = (k0i <= qrow) && (qrow - k0i < WINDOW);
            const bool v1 = (k1i <= qrow) && (qrow - k1i < WINDOW);
            s0 = v0 ? s0 : -1.0e9f;
            s1 = v1 ? s1 : -1.0e9f;

            float rm = fmaxf(s0, s1);
            rm = fmaxf(rm, __shfl_xor(rm, 1, 32));
            rm = fmaxf(rm, __shfl_xor(rm, 2, 32));
            rm = fmaxf(rm, __shfl_xor(rm, 4, 32));
            rm = fmaxf(rm, __shfl_xor(rm, 8, 32));

            const float mnew  = fmaxf(mrun[r], rm);
            const float corr  = __expf(mrun[r] - mnew);
            const float p0 = __expf(s0 - mnew);
            const float p1 = __expf(s1 - mnew);
            float rs = p0 + p1;
            rs += __shfl_xor(rs, 1, 32);
            rs += __shfl_xor(rs, 2, 32);
            rs += __shfl_xor(rs, 4, 32);
            rs += __shfl_xor(rs, 8, 32);

            lrun[r] = lrun[r] * corr + rs;
            mrun[r] = mnew;
#pragma unroll
            for (int j = 0; j < 8; ++j) o[j][r] *= corr;

            Ps[wave][r + 8 * half][l15]      = (__bf16)p0;
            Ps[wave][r + 8 * half][16 + l15] = (__bf16)p1;
        }
        asm volatile("s_wait_dscnt 0" ::: "memory");

        // ---- P (A-layout 16x32) from LDS ----
        FragAB pf;
        const __bf16* pp = &Ps[wave][l15][half * 16];
        pf.h[0] = *(const bf16x8*)(pp);
        pf.h[1] = *(const bf16x8*)(pp + 8);

        // ---- O += P @ V : 8 head-dim tiles; V B-operand from transposed vt ----
#pragma unroll
        for (int j = 0; j < 8; ++j) {
            FragAB vf;
            const __bf16* vp = Vt + (size_t)(j * 16 + l15) * SEQL + kk + half * 16;
            vf.h[0] = *(const bf16x8*)(vp);
            vf.h[1] = *(const bf16x8*)(vp + 8);
            o[j] = __builtin_amdgcn_wmma_f32_16x16x32_bf16(false, pf.v, false, vf.v, (short)0, o[j], false, false);
        }
    }

    // ---- epilogue: normalize and scatter to [B, S, D] bf16 ----
#pragma unroll
    for (int r = 0; r < 8; ++r) {
        const int qrow = q0 + r + 8 * half;
        const float inv = 1.0f / lrun[r];
        const size_t base = ((size_t)(b * SEQL + qrow)) * D_MODEL + h * HEAD_DIM;
#pragma unroll
        for (int j = 0; j < 8; ++j)
            out[base + j * 16 + l15] = (__bf16)(o[j][r] * inv);
    }
}

// ---------------------------------------------------------------------------
extern "C" void kernel_launch(void* const* d_in, const int* in_sizes, int n_in,
                              void* d_out, int out_size, void* d_ws, size_t ws_size,
                              hipStream_t stream) {
    const float* x    = (const float*)d_in[0];
    const float* fcos = (const float*)d_in[1];
    const float* fsin = (const float*)d_in[2];
    // d_in[3] = mask (unused; causal+window applied analytically)
    const float* wq = (const float*)d_in[4];
    const float* wk = (const float*)d_in[5];
    const float* wv = (const float*)d_in[6];
    const float* wo = (const float*)d_in[7];

    char* ws = (char*)d_ws;
    const size_t MB = 1024 * 1024;
    __bf16* xb   = (__bf16*)(ws);              // 16MB; reused for attention output
    __bf16* wqb  = (__bf16*)(ws + 16  * MB);   // 8MB each
    __bf16* wkb  = (__bf16*)(ws + 24  * MB);
    __bf16* wvb  = (__bf16*)(ws + 32  * MB);
    __bf16* wob  = (__bf16*)(ws + 40  * MB);
    __bf16* qlin = (__bf16*)(ws + 48  * MB);   // 16MB each
    __bf16* klin = (__bf16*)(ws + 64  * MB);
    __bf16* vlin = (__bf16*)(ws + 80  * MB);
    __bf16* qhh  = (__bf16*)(ws + 96  * MB);
    __bf16* khh  = (__bf16*)(ws + 112 * MB);
    __bf16* vtt  = (__bf16*)(ws + 128 * MB);   // total 144MB

    const int nx = BATCH * SEQL * D_MODEL;     // 8,388,608
    const int nw = D_MODEL * D_MODEL;          // 4,194,304
    cvt_f32_bf16<<<(nx + 255) / 256, 256, 0, stream>>>(x,  xb,  nx);
    cvt_f32_bf16<<<(nw + 255) / 256, 256, 0, stream>>>(wq, wqb, nw);
    cvt_f32_bf16<<<(nw + 255) / 256, 256, 0, stream>>>(wk, wkb, nw);
    cvt_f32_bf16<<<(nw + 255) / 256, 256, 0, stream>>>(wv, wvb, nw);
    cvt_f32_bf16<<<(nw + 255) / 256, 256, 0, stream>>>(wo, wob, nw);

    dim3 gg(D_MODEL / 128, MROWS / 128);       // (16, 32)
    gemm_bf16<false><<<gg, 256, 0, stream>>>(xb, wqb, qlin, nullptr, MROWS, D_MODEL, D_MODEL);
    gemm_bf16<false><<<gg, 256, 0, stream>>>(xb, wkb, klin, nullptr, MROWS, D_MODEL, D_MODEL);
    gemm_bf16<false><<<gg, 256, 0, stream>>>(xb, wvb, vlin, nullptr, MROWS, D_MODEL, D_MODEL);

    rope_relayout<<<BATCH * SEQL, 256, 0, stream>>>(qlin, klin, vlin, fcos, fsin, qhh, khh, vtt);

    attention_kernel<<<BATCH * NUM_HEADS * (SEQL / 128), 256, 0, stream>>>(qhh, khh, vtt, xb);

    gemm_bf16<true><<<gg, 256, 0, stream>>>(xb, wob, nullptr, (float*)d_out, MROWS, D_MODEL, D_MODEL);
}
